// negLogLikelihood_per_sample_for_splitting_45990509805832
// MI455X (gfx1250) — compile-verified, matrix operand order checked by reading, not run on hardware
//
#include <hip/hip_runtime.h>
#include <hip/hip_bf16.h>

typedef __attribute__((ext_vector_type(16))) _Float16 v16h;
typedef __attribute__((ext_vector_type(8)))  _Float16 v8h;
typedef __attribute__((ext_vector_type(8)))  float    v8f;
typedef __attribute__((ext_vector_type(8)))  float    v8f32;

#define N_SAMP  4096
#define M_REF   16384
#define K_TILE  4096          // LDS tile of reference subjects
#define WAVES   4             // wave32s per block (smaller block -> 64 WGPs busy)
#define BLOCK   (WAVES * 32)

// One wave computes partial_sum for 16 samples via v_wmma_f32_16x16x32_f16:
//   A[m][k] = (refer_time[k] >= time_m) ? (f16)exp(refer_pred[k]) : 0
//   B       = ones  ->  D[m][n] = partial_sum[m]  (all columns identical)
__global__ __launch_bounds__(BLOCK)
void cox_nll_wmma(const float* __restrict__ pred,
                  const int*   __restrict__ targets,
                  const float* __restrict__ time_,
                  const float* __restrict__ refer_pred,
                  const float* __restrict__ refer_time,
                  float* __restrict__ out)
{
    __shared__ alignas(32) float    s_rt[K_TILE];     // 16 KB
    __shared__ alignas(32) _Float16 s_h [K_TILE];     // 8 KB
    __shared__ float                s_ps[WAVES * 16]; // per-wave row sums

    const int tid  = threadIdx.x;
    const int lane = tid & 31;
    const int wave = tid >> 5;
    const int m    = lane & 15;   // matrix row (sample within tile)
    const int side = lane >> 4;   // K-group half per CDNA5 16-bit A layout
    const int koff = side * 8;

    const int   tile = blockIdx.x * WAVES + wave;   // 16-sample tile id
    const int   i    = tile * 16 + m;               // global sample index
    const float tm   = time_[i];

    v16h ones;
    #pragma unroll
    for (int e = 0; e < 16; ++e) ones[e] = (_Float16)1.0f;

    const _Float16 zh = (_Float16)0.0f;

    v8f acc = {};

    for (int kt = 0; kt < M_REF; kt += K_TILE) {
        // Cooperative fill: refer_time + f16 hazard (exp fused into the load).
        for (int k = tid; k < K_TILE; k += BLOCK) {
            s_rt[k] = refer_time[kt + k];
            s_h [k] = (_Float16)__expf(refer_pred[kt + k]);
            if (kt + K_TILE < M_REF) {   // prefetch next tile -> global_prefetch_b8
                __builtin_prefetch(&refer_time[kt + K_TILE + k], 0, 1);
                __builtin_prefetch(&refer_pred[kt + K_TILE + k], 0, 1);
            }
        }
        __syncthreads();

        // Software-pipelined K loop: issue next chunk's ds_load_b128s into
        // fresh registers BEFORE current chunk's select+WMMA, so LDS latency
        // overlaps VALU/WMMA instead of stalling on s_wait_dscnt each load.
        v8f32 r0 = *reinterpret_cast<const v8f32*>(&s_rt[koff]);
        v8f32 r1 = *reinterpret_cast<const v8f32*>(&s_rt[koff + 16]);
        v8h   h0 = *reinterpret_cast<const v8h*>(&s_h[koff]);
        v8h   h1 = *reinterpret_cast<const v8h*>(&s_h[koff + 16]);

        #pragma unroll 2
        for (int kb = 0; kb < K_TILE; kb += 32) {
            const int kn = (kb + 32) & (K_TILE - 1);  // wraps on last iter (no branch)
            const int ko = kn + koff;
            const v8f32 nr0 = *reinterpret_cast<const v8f32*>(&s_rt[ko]);
            const v8f32 nr1 = *reinterpret_cast<const v8f32*>(&s_rt[ko + 16]);
            const v8h   nh0 = *reinterpret_cast<const v8h*>(&s_h[ko]);
            const v8h   nh1 = *reinterpret_cast<const v8h*>(&s_h[ko + 16]);

            v16h a;
            #pragma unroll
            for (int e = 0; e < 8; ++e)      // elements 0..7  -> K = kb+koff+e
                a[e]     = (r0[e] >= tm) ? h0[e] : zh;
            #pragma unroll
            for (int e = 0; e < 8; ++e)      // elements 8..15 -> K = kb+koff+16+e
                a[8 + e] = (r1[e] >= tm) ? h1[e] : zh;

            acc = __builtin_amdgcn_wmma_f32_16x16x32_f16(
                      /*neg_a=*/false, a, /*neg_b=*/false, ones,
                      /*c_mod=*/(short)0, acc, /*reuse_a=*/false, /*reuse_b=*/false);

            r0 = nr0; r1 = nr1; h0 = nh0; h1 = nh1;
        }
        __syncthreads();   // protect LDS before next tile overwrites it
    }

    // D layout: VGPR v = rows {v (lanes 0-15), v+8 (lanes 16-31)}; cols = lanes,
    // all equal. Lane 0 -> rows 0..7, lane 16 -> rows 8..15.
    if (lane == 0 || lane == 16) {
        #pragma unroll
        for (int v = 0; v < 8; ++v)
            s_ps[wave * 16 + side * 8 + v] = acc[v];
    }
    __syncthreads();

    if (lane < 16) {
        const float ps    = s_ps[wave * 16 + lane];
        const float p     = pred[i];
        const float total = ps + __expf(p);          // + self hazard
        out[i] = -(p - __logf(total)) * (float)targets[i];
    }
}

extern "C" void kernel_launch(void* const* d_in, const int* in_sizes, int n_in,
                              void* d_out, int out_size, void* d_ws, size_t ws_size,
                              hipStream_t stream)
{
    (void)in_sizes; (void)n_in; (void)d_ws; (void)ws_size; (void)out_size;
    const float* pred       = (const float*)d_in[0];  // prediction        [N,1]
    const int*   targets    = (const int*)  d_in[1];  // targets           [N]
    const float* time_      = (const float*)d_in[2];  // time              [N]
    const float* refer_pred = (const float*)d_in[3];  // refer_prediction  [M,1]
    const float* refer_time = (const float*)d_in[4];  // refer_time        [M]
    float*       out        = (float*)d_out;          // [1,N] -> N floats

    const int blocks = N_SAMP / (16 * WAVES);         // 4096 / 64 = 64
    cox_nll_wmma<<<blocks, BLOCK, 0, stream>>>(pred, targets, time_,
                                               refer_pred, refer_time, out);
}
